// ResNetCustom_85761906966889
// MI455X (gfx1250) — matrix operand order repaired and checked by reading, not compiled, not run on hardware
//
#include <hip/hip_runtime.h>
#include <hip/hip_bf16.h>

// ---------------------------------------------------------------------------
// Types for CDNA5 WMMA (wave32): v16bf = A/B fragment, v8f = C/D fragment
// ---------------------------------------------------------------------------
typedef __attribute__((ext_vector_type(16))) __bf16 v16bf;
typedef __attribute__((ext_vector_type(8)))  float  v8f;

union ABFrag { uint4 q[2]; v16bf v; };

__device__ __forceinline__ unsigned int f2bf1(float f) {
    unsigned int u = __float_as_uint(f);
    return (u + 0x7FFFu + ((u >> 16) & 1u)) >> 16;   // round-to-nearest-even
}

// ---------------------------------------------------------------------------
// Weight convert + transpose: W[k][cin][cout] (f32) -> Wt[k][cout][cinp] (bf16)
// cinp = max(cin,32), zero padded so the K loop always runs in chunks of 32.
// ---------------------------------------------------------------------------
__global__ void wcvt(const float* __restrict__ w, unsigned short* __restrict__ wt,
                     int kvol, int cin, int cinp, int cout) {
    int i = blockIdx.x * blockDim.x + threadIdx.x;
    int total = kvol * cout * cinp;
    if (i >= total) return;
    int c   = i % cinp;
    int rem = i / cinp;
    int n   = rem % cout;
    int k   = rem / cout;
    unsigned short v = 0;
    if (c < cin) v = (unsigned short)f2bf1(w[((size_t)k * cin + c) * cout + n]);
    wt[i] = v;
}

// ---------------------------------------------------------------------------
// Pad + convert input features: f32 [npts][4] -> bf16 [npts][32] (zero padded)
// ---------------------------------------------------------------------------
__global__ void featpad(const float* __restrict__ f, unsigned short* __restrict__ o,
                        int npts) {
    int i = blockIdx.x * blockDim.x + threadIdx.x;
    if (i >= npts * 32) return;
    int c = i & 31;
    int r = i >> 5;
    o[i] = (c < 4) ? (unsigned short)f2bf1(f[r * 4 + c]) : 0;
}

// ---------------------------------------------------------------------------
// Per-channel instance-norm statistics: one block per channel.
// stats[c] = mean, stats[C+c] = rsqrt(var + eps)
// ---------------------------------------------------------------------------
__global__ __launch_bounds__(256) void colstats(const float* __restrict__ x,
                                                int npts, int C,
                                                float* __restrict__ stats) {
    int c = blockIdx.x;
    float s = 0.f, ss = 0.f;
    for (int r = threadIdx.x; r < npts; r += 256) {
        float v = x[(size_t)r * C + c];
        s += v; ss += v * v;
    }
    __shared__ float sh[512];
    sh[threadIdx.x] = s;
    sh[256 + threadIdx.x] = ss;
    __syncthreads();
    for (int o = 128; o > 0; o >>= 1) {
        if (threadIdx.x < o) {
            sh[threadIdx.x]       += sh[threadIdx.x + o];
            sh[256 + threadIdx.x] += sh[256 + threadIdx.x + o];
        }
        __syncthreads();
    }
    if (threadIdx.x == 0) {
        float inv = 1.0f / (float)npts;
        float mu  = sh[0] * inv;
        float var = sh[256] * inv - mu * mu;
        stats[c]     = mu;
        stats[C + c] = rsqrtf(var + 1e-5f);
    }
}

// ---------------------------------------------------------------------------
// Fused instance-norm (scale/shift) + optional residual + LeakyReLU(0.01).
// Emits fp32 (optional, may alias res: elementwise 1:1 -> safe) and/or a bf16
// mirror that feeds the next WMMA conv.
// ---------------------------------------------------------------------------
__global__ __launch_bounds__(256) void norm_act(const float* __restrict__ x,
                                                const float* __restrict__ stats,
                                                const float* __restrict__ g,
                                                const float* __restrict__ b,
                                                const float* res, float* y,
                                                unsigned short* __restrict__ ybf,
                                                int npts, int C) {
    size_t i = (size_t)blockIdx.x * blockDim.x + threadIdx.x;
    size_t total = (size_t)npts * C;
    if (i >= total) return;
    int c = (int)(i % C);
    float v = (x[i] - stats[c]) * stats[C + c] * g[c] + b[c];
    if (res) v += res[i];
    v = (v >= 0.f) ? v : 0.01f * v;
    if (y)   y[i]   = v;
    if (ybf) ybf[i] = (unsigned short)f2bf1(v);
}

// ---------------------------------------------------------------------------
// Gather-GEMM sparse conv, bf16 WMMA with f32 accumulate.
//   out[n,:] = sum_{k<KVOL} xbf[nbr[n,k],:] @ W[k]   (KVOL=27; KVOL=1: dense)
// xbf: bf16 activations, row stride CINP.  Wt: bf16 [KVOL][COUT][CINP].
// Block: 256 threads (8 waves) -> 32 rows x COUT cols.
// Per k-offset: one barrier pair stages the full 32 x CINP gathered tile in
// LDS (padded stride, conflict-free ds_load_b128); then the kc loop is pure
// ds_load + global_load + v_wmma_f32_16x16x32_bf16 with register accumulators.
// Wave w: fixed M-half (w>>2), columns nt = (w&3) + 4j  (A frag loaded once
// per kc, B frag once per owned column tile).
// ---------------------------------------------------------------------------
template<int CIN, int COUT, int KVOL>
__global__ __launch_bounds__(256)
void spconv_wmma(const unsigned short* __restrict__ xbf,
                 const int* __restrict__ nbr,
                 const unsigned short* __restrict__ Wt,
                 float* __restrict__ out, int npts) {
    constexpr int CINP = (CIN < 32) ? 32 : CIN;   // padded K extent
    constexpr int KC   = CINP / 32;               // K chunks of 32
    constexpr int NT   = COUT / 16;               // 16-wide column tiles
    constexpr int NJ   = (NT + 3) / 4;            // column tiles per wave
    constexpr int LDSS = CINP + 8;                // LDS row stride (bf16 units)

    __shared__ __align__(16) unsigned short As[32 * LDSS];

    const int tid   = threadIdx.x;
    const int lane  = tid & 31;
    const int wave  = tid >> 5;
    const int mSubW = wave >> 2;                  // 0..1 : M-half owned
    const int ntW   = wave & 3;                   // base column tile
    const int rowbase = blockIdx.x * 32;

    v8f zero = {0.f, 0.f, 0.f, 0.f, 0.f, 0.f, 0.f, 0.f};
    v8f acc[NJ];
    for (int j = 0; j < NJ; ++j) acc[j] = zero;

    // gather mapping: thread -> (row, 8-channel group)
    const int gr   = tid >> 3;                    // 0..31
    const int gc8  = (tid & 7) * 8;               // 0..56
    const int grow = rowbase + gr;

    const int m   = lane & 15;
    const int kb  = (lane >> 4) * 8;              // A frag K sub-base
    const int kb2 = (lane >> 4) * 16;             // unused (B frag also uses kb)

    for (int k = 0; k < KVOL; ++k) {
        int src = -1;
        if (grow < npts) src = (KVOL == 27) ? nbr[grow * 27 + k] : grow;

        __syncthreads();   // protect previous k's A-frag reads
        #pragma unroll
        for (int cc = 0; cc < CINP; cc += 64) {
            int col = cc + gc8;
            if (col < CINP) {
                uint4 v = make_uint4(0u, 0u, 0u, 0u);
                if (src >= 0)
                    v = *(const uint4*)(xbf + (size_t)src * CINP + col);
                *(uint4*)&As[gr * LDSS + col] = v;
            }
        }
        __syncthreads();

        for (int kc = 0; kc < KC; ++kc) {
            // A fragment (ISA 16-bit A 16x32 layout): lanes 0-15 hold
            // K=kb..kb+7 (VGPR0..3) and K=kb+16..kb+23 (VGPR4..7)
            const int abase = (mSubW * 16 + m) * LDSS + kc * 32 + kb;
            ABFrag a;
            a.q[0] = *(const uint4*)&As[abase];
            a.q[1] = *(const uint4*)&As[abase + 16];

            #pragma unroll
            for (int j = 0; j < NJ; ++j) {
                int nt = ntW + j * 4;
                if (nt >= NT) continue;
                // B fragment: lane = column, bf16 of
                // Wt[k][nt*16+m][kc*32 + kb ..+7] and [kc*32 + kb+16 ..+23]
                const uint4* bp = (const uint4*)(Wt +
                    ((size_t)(k * COUT + nt * 16 + m)) * CINP + kc * 32 + kb);
                ABFrag bb;
                bb.q[0] = bp[0];
                bb.q[1] = bp[2];
                acc[j] = __builtin_amdgcn_wmma_f32_16x16x32_bf16(
                    false, a.v, false, bb.v, (short)0, acc[j], false, false);
            }
        }
    }

    // C/D layout: lane VGPR r -> row (lane>>4)*8 + r, col = lane&15
    const int mrow = rowbase + mSubW * 16 + (lane >> 4) * 8;
    #pragma unroll
    for (int j = 0; j < NJ; ++j) {
        int nt = ntW + j * 4;
        if (nt >= NT) continue;
        int col = nt * 16 + (lane & 15);
        #pragma unroll
        for (int r = 0; r < 8; ++r) {
            int row = mrow + r;
            if (row < npts) out[(size_t)row * COUT + col] = acc[j][r];
        }
    }
}

// ---------------------------------------------------------------------------
// Host-side helpers
// ---------------------------------------------------------------------------
template<int CIN, int COUT, int KVOL>
static inline void launch_conv(const unsigned short* xbf, const int* nbr,
                               const unsigned short* wt, float* out,
                               int npts, hipStream_t s) {
    spconv_wmma<CIN, COUT, KVOL>
        <<<dim3((unsigned)((npts + 31) / 32)), dim3(256), 0, s>>>(xbf, nbr, wt, out, npts);
}

extern "C" void kernel_launch(void* const* d_in, const int* in_sizes, int n_in,
                              void* d_out, int out_size, void* d_ws, size_t ws_size,
                              hipStream_t stream) {
    (void)n_in; (void)out_size; (void)ws_size;

    const float* feats = (const float*)d_in[0];
    const int*   nbr   = (const int*)d_in[1];
    const int npts = in_sizes[0] / 4;              // IN_CH = 4

    // ---- workspace layout -------------------------------------------------
    float* bufT  = (float*)d_ws;                          // conv out   (npts*256 f32)
    float* bufP  = bufT + (size_t)npts * 256;             // fp32 master (residual)
    float* stats = bufP + (size_t)npts * 256;             // 512 f32
    unsigned short* bf0 = (unsigned short*)(stats + 512); // bf16 master (npts*256)
    unsigned short* bf1 = bf0 + (size_t)npts * 256;       // bf16 intermediate
    unsigned short* fbf = bf1 + (size_t)npts * 256;       // padded feats (npts*32)
    unsigned short* warena = fbf + (size_t)npts * 32;     // bf16 weight arena

    // ---- param walk (setup_inputs dict order) -----------------------------
    int pi = 2;
    auto nextp = [&]() { return (const float*)d_in[pi++]; };

    size_t woff = 0;
    auto cvtw = [&](const float* w, int kvol, int cin, int cout) -> const unsigned short* {
        int cinp = (cin < 32) ? 32 : cin;
        unsigned short* dst = warena + woff;
        size_t total = (size_t)kvol * cout * cinp;
        woff += total;
        wcvt<<<dim3((unsigned)((total + 255) / 256)), dim3(256), 0, stream>>>(
            w, dst, kvol, cin, cinp, cout);
        return dst;
    };

    struct BlockP {
        const unsigned short* w1; const float *g1, *b1;
        const unsigned short* w2; const float *g2, *b2;
    };
    auto getblock = [&](int c) -> BlockP {
        BlockP bp;
        bp.w1 = cvtw(nextp(), 27, c, c); bp.g1 = nextp(); bp.b1 = nextp();
        bp.w2 = cvtw(nextp(), 27, c, c); bp.g2 = nextp(); bp.b2 = nextp();
        return bp;
    };

    const unsigned short* conv1_w = cvtw(nextp(), 27, 4, 64);
    const float* conv1_g = nextp(); const float* conv1_b = nextp();

    BlockP L1[3]; for (int i = 0; i < 3; ++i) L1[i] = getblock(64);
    const unsigned short* proj1_w = cvtw(nextp(), 1, 64, 128);
    const float* proj1_g = nextp(); const float* proj1_b = nextp();

    BlockP L2[3]; for (int i = 0; i < 3; ++i) L2[i] = getblock(128);
    const unsigned short* proj2_w = cvtw(nextp(), 1, 128, 256);
    const float* proj2_g = nextp(); const float* proj2_b = nextp();

    BlockP L3[3]; for (int i = 0; i < 3; ++i) L3[i] = getblock(256);
    const unsigned short* proj3_w = cvtw(nextp(), 1, 256, 96);
    const float* proj3_g = nextp(); const float* proj3_b = nextp();

    BlockP L4 = getblock(96);

    // ---- elementwise launchers -------------------------------------------
    auto stat = [&](const float* t, int C) {
        colstats<<<dim3((unsigned)C), dim3(256), 0, stream>>>(t, npts, C, stats);
    };
    auto nact = [&](const float* t, const float* g, const float* b,
                    const float* res, float* y, unsigned short* ybf, int C) {
        size_t total = (size_t)npts * C;
        norm_act<<<dim3((unsigned)((total + 255) / 256)), dim3(256), 0, stream>>>(
            t, stats, g, b, res, y, ybf, npts, C);
    };

    // ---- stem: feats -> bf16 pad, conv1 -> inorm -> lrelu ------------------
    featpad<<<dim3((unsigned)((npts * 32 + 255) / 256)), dim3(256), 0, stream>>>(
        feats, fbf, npts);
    launch_conv<4, 64, 27>(fbf, nbr, conv1_w, bufT, npts, stream);
    stat(bufT, 64);
    nact(bufT, conv1_g, conv1_b, nullptr, bufP, bf0, 64);
    // masters: bufP (fp32), bf0 (bf16)

    // ---- layer1 (64) ------------------------------------------------------
    for (int i = 0; i < 3; ++i) {
        launch_conv<64, 64, 27>(bf0, nbr, L1[i].w1, bufT, npts, stream);
        stat(bufT, 64); nact(bufT, L1[i].g1, L1[i].b1, nullptr, nullptr, bf1, 64);
        launch_conv<64, 64, 27>(bf1, nbr, L1[i].w2, bufT, npts, stream);
        stat(bufT, 64); nact(bufT, L1[i].g2, L1[i].b2, bufP, bufP, bf0, 64);
    }
    launch_conv<64, 128, 1>(bf0, nbr, proj1_w, bufT, npts, stream);
    stat(bufT, 128); nact(bufT, proj1_g, proj1_b, nullptr, bufP, bf0, 128);

    // ---- layer2 (128) -----------------------------------------------------
    for (int i = 0; i < 3; ++i) {
        launch_conv<128, 128, 27>(bf0, nbr, L2[i].w1, bufT, npts, stream);
        stat(bufT, 128); nact(bufT, L2[i].g1, L2[i].b1, nullptr, nullptr, bf1, 128);
        launch_conv<128, 128, 27>(bf1, nbr, L2[i].w2, bufT, npts, stream);
        stat(bufT, 128); nact(bufT, L2[i].g2, L2[i].b2, bufP, bufP, bf0, 128);
    }
    launch_conv<128, 256, 1>(bf0, nbr, proj2_w, bufT, npts, stream);
    stat(bufT, 256); nact(bufT, proj2_g, proj2_b, nullptr, bufP, bf0, 256);

    // ---- layer3 (256) -----------------------------------------------------
    for (int i = 0; i < 3; ++i) {
        launch_conv<256, 256, 27>(bf0, nbr, L3[i].w1, bufT, npts, stream);
        stat(bufT, 256); nact(bufT, L3[i].g1, L3[i].b1, nullptr, nullptr, bf1, 256);
        launch_conv<256, 256, 27>(bf1, nbr, L3[i].w2, bufT, npts, stream);
        stat(bufT, 256); nact(bufT, L3[i].g2, L3[i].b2, bufP, bufP, bf0, 256);
    }
    launch_conv<256, 96, 1>(bf0, nbr, proj3_w, bufT, npts, stream);
    stat(bufT, 96); nact(bufT, proj3_g, proj3_b, nullptr, bufP, bf0, 96);

    // ---- layer4: single basic block (96), final result -> d_out -----------
    launch_conv<96, 96, 27>(bf0, nbr, L4.w1, bufT, npts, stream);
    stat(bufT, 96); nact(bufT, L4.g1, L4.b1, nullptr, nullptr, bf1, 96);
    launch_conv<96, 96, 27>(bf1, nbr, L4.w2, bufT, npts, stream);
    stat(bufT, 96); nact(bufT, L4.g2, L4.b2, bufP, (float*)d_out, nullptr, 96);
}
